// LlamaAttention_68178310856909
// MI455X (gfx1250) — matrix operand order repaired and checked by reading, not compiled, not run on hardware
//
#include <hip/hip_runtime.h>
#include <hip/hip_bf16.h>

// ---------------------------------------------------------------------------
// Types
// ---------------------------------------------------------------------------
typedef __bf16 bf16;
typedef __attribute__((ext_vector_type(16))) __bf16        v16bf;
typedef __attribute__((ext_vector_type(8)))  float         v8f;
typedef __attribute__((ext_vector_type(4)))  unsigned int  u32x4;
typedef __attribute__((ext_vector_type(4)))  unsigned int  uint32x4;
typedef __attribute__((ext_vector_type(8)))  int           int32x8;
typedef __attribute__((ext_vector_type(4)))  int           int32x4;

// ---------------------------------------------------------------------------
// CDNA5 feature probes (fall back to the plain path if builtins are absent)
// ---------------------------------------------------------------------------
#if __has_builtin(__builtin_amdgcn_global_load_async_to_lds_b128)
#define HAVE_ASYNC 1
#else
#define HAVE_ASYNC 0
#endif

#if __has_builtin(__builtin_amdgcn_tensor_load_to_lds) && __has_builtin(__builtin_amdgcn_s_wait_tensorcnt)
#define HAVE_TDM 1
#else
#define HAVE_TDM 0
#endif

#if HAVE_ASYNC
typedef __attribute__((address_space(1))) int32x4 gas_int4;
typedef __attribute__((address_space(3))) int32x4 lds_int4;

static __device__ __forceinline__ gas_int4* to_global4(const void* p) {
    return (gas_int4*)(uintptr_t)p;
}
static __device__ __forceinline__ lds_int4* to_lds4(void* p) {
    // flat shared address: low 32 bits are the LDS byte offset
    return (lds_int4*)(uintptr_t)(uint32_t)(uintptr_t)p;
}
#endif

static __device__ __forceinline__ void wait_async0() {
#if HAVE_ASYNC
#if __has_builtin(__builtin_amdgcn_s_wait_asynccnt)
    __builtin_amdgcn_s_wait_asynccnt(0);
#else
    asm volatile("s_wait_asynccnt 0x0" ::: "memory");
#endif
#endif
}

static __device__ __forceinline__ v8f zero8() {
    return (v8f){0.f, 0.f, 0.f, 0.f, 0.f, 0.f, 0.f, 0.f};
}

// WMMA: D = A(16x32 bf16) * B(32x16 bf16) + C(16x16 f32)
#define WMMA_BF16(a, b, c) \
    __builtin_amdgcn_wmma_f32_16x16x32_bf16(false, (a), false, (b), (short)0, (c), false, false)

// A-operand fragment (16x32, M x K). lane&15 = M row;
// halves 0..7 -> K = 8*(lane>>4)+0..7 ; halves 8..15 -> K = +16.
static __device__ __forceinline__ v16bf frag_ld_A(const bf16* p0, int ld, int lane) {
    const bf16* p = p0 + (lane & 15) * ld + ((lane >> 4) << 3);
    union { v16bf v; u32x4 q[2]; } u;
    u.q[0] = *(const u32x4*)(p);
    u.q[1] = *(const u32x4*)(p + 16);
    return u.v;
}

// B-operand fragment (32x16, K x N). lane&15 = N col; halves 0..7 -> K = 8*(lane>>4)+0..7 ; halves 8..15 -> K = +16.
static __device__ __forceinline__ v16bf frag_ld_B(const bf16* p0, int ld, int lane) {
    const bf16* p = p0 + (lane & 15) * ld + ((lane >> 4) << 3);
    union { v16bf v; u32x4 q[2]; } u;
    u.q[0] = *(const u32x4*)(p);
    u.q[1] = *(const u32x4*)(p + 16);
    return u.v;
}

// ---------------------------------------------------------------------------
// f32 -> bf16 conversion (bandwidth-trivial)
// ---------------------------------------------------------------------------
__global__ void f32_to_bf16_kernel(const float* __restrict__ in, bf16* __restrict__ out, long n) {
    long i = (long)blockIdx.x * blockDim.x + threadIdx.x;
    long stride = (long)gridDim.x * blockDim.x;
    for (; i < n; i += stride) out[i] = (bf16)in[i];
}

// ---------------------------------------------------------------------------
// GEMM: C[M,N] = A[M,K] @ B[N,K]^T  (both operands K-major bf16, f32 accum)
// Block tile 128x128, BK=64, double-buffered LDS, async global->LDS staging.
// 8 waves as 2(M) x 4(N); each wave: 4x2 WMMA tiles x 2 K-chunks per step.
// ---------------------------------------------------------------------------
template <bool OUT_BF16, bool WITH_BIAS>
__global__ __launch_bounds__(256) void gemm_bf16_kernel(
    const bf16* __restrict__ A, const bf16* __restrict__ B,
    const float* __restrict__ bias,
    float* __restrict__ Cf, bf16* __restrict__ Cb,
    int M, int N, int K)
{
    __shared__ __align__(16) bf16 As[2][128][64];   // 32 KB
    __shared__ __align__(16) bf16 Bs[2][128][64];   // 32 KB

    const int t    = threadIdx.x;
    const int lane = t & 31;
    const int w    = t >> 5;
    const int wm   = (w >> 2) * 64;
    const int wn   = (w & 3) * 32;
    const int bm   = blockIdx.y * 128;
    const int bn   = blockIdx.x * 128;

    // stage one 128x64 tile pair into LDS buffer `bufi` (1024 chunks of 16B each)
    auto stage = [&](int bufi, int k0) {
#pragma unroll
        for (int i = 0; i < 4; ++i) {
            int c  = t + i * 256;
            int r  = c >> 3;
            int cc = (c & 7) << 3;
            const bf16* ga = A + (size_t)(bm + r) * K + k0 + cc;
            const bf16* gb = B + (size_t)(bn + r) * K + k0 + cc;
#if HAVE_ASYNC
            __builtin_amdgcn_global_load_async_to_lds_b128(
                to_global4(ga), to_lds4(&As[bufi][r][cc]), 0, 0);
            __builtin_amdgcn_global_load_async_to_lds_b128(
                to_global4(gb), to_lds4(&Bs[bufi][r][cc]), 0, 0);
#else
            *(u32x4*)&As[bufi][r][cc] = *(const u32x4*)ga;
            *(u32x4*)&Bs[bufi][r][cc] = *(const u32x4*)gb;
#endif
        }
    };

    v8f acc[4][2];
#pragma unroll
    for (int i = 0; i < 4; ++i)
#pragma unroll
        for (int j = 0; j < 2; ++j) acc[i][j] = zero8();

    stage(0, 0);
    int buf = 0;
    for (int k0 = 0; k0 < K; k0 += 64) {
        wait_async0();        // my async stores into `buf` complete
        __syncthreads();      // everyone's tile in LDS; previous reads retired
        if (k0 + 64 < K) stage(buf ^ 1, k0 + 64);   // prefetch next tile pair

#pragma unroll
        for (int kk = 0; kk < 2; ++kk) {
            v16bf a[4], b[2];
#pragma unroll
            for (int i = 0; i < 4; ++i)
                a[i] = frag_ld_A(&As[buf][wm + i * 16][kk * 32], 64, lane);
#pragma unroll
            for (int j = 0; j < 2; ++j)
                b[j] = frag_ld_B(&Bs[buf][wn + j * 16][kk * 32], 64, lane);
#pragma unroll
            for (int i = 0; i < 4; ++i)
#pragma unroll
                for (int j = 0; j < 2; ++j)
                    acc[i][j] = WMMA_BF16(a[i], b[j], acc[i][j]);
        }
        buf ^= 1;
    }

    // epilogue: C element (r + 8*(lane>>4), lane&15) in VGPR r
    const int rhalf = (lane >> 4) << 3;
    const int cl    = lane & 15;
#pragma unroll
    for (int i = 0; i < 4; ++i) {
#pragma unroll
        for (int j = 0; j < 2; ++j) {
            int col = bn + wn + j * 16 + cl;
            float bv = 0.f;
            if constexpr (WITH_BIAS) bv = bias[col];
#pragma unroll
            for (int r = 0; r < 8; ++r) {
                int row = bm + wm + i * 16 + r + rhalf;
                float v = acc[i][j][r] + bv;
                if constexpr (OUT_BF16) Cb[(size_t)row * N + col] = (bf16)v;
                else                    Cf[(size_t)row * N + col] = v;
            }
        }
    }
}

// ---------------------------------------------------------------------------
// RoPE on bf16 Q/K laid out [N, S, nh, dh]; math in f32
// ---------------------------------------------------------------------------
__global__ void rope_kernel(bf16* __restrict__ Q, bf16* __restrict__ K,
                            const int* __restrict__ pos_ids, long total) {
    const float coef = 0.14391156507f;  // ln(10000)/64
    long i = (long)blockIdx.x * blockDim.x + threadIdx.x;
    long stride = (long)gridDim.x * blockDim.x;
    for (; i < total; i += stride) {
        int j    = (int)(i & 63);
        long rest = i >> 6;                 // (n*S + s)*16 + h
        int s    = (int)((rest >> 4) & 2047);
        long base = rest * 128;
        float p   = (float)pos_ids[s];
        float inv = __expf(-(float)j * coef);
        float ang = p * inv;
        float c   = __cosf(ang);
        float sn  = __sinf(ang);
        float q1 = (float)Q[base + j], q2 = (float)Q[base + 64 + j];
        float k1 = (float)K[base + j], k2 = (float)K[base + 64 + j];
        Q[base + j]      = (bf16)(q1 * c - q2 * sn);
        Q[base + 64 + j] = (bf16)(q1 * sn + q2 * c);
        K[base + j]      = (bf16)(k1 * c - k2 * sn);
        K[base + 64 + j] = (bf16)(k1 * sn + k2 * c);
    }
}

// ---------------------------------------------------------------------------
// Flash attention, causal. Grid: (S/128, nh, N). Block: 256 (8 waves).
// K tile staged via TDM (tensor_load_to_lds) when available; V staged
// transposed by vector lanes in parallel with the TDM transfer.
// ---------------------------------------------------------------------------
__global__ __launch_bounds__(256) void flash_attn_kernel(
    const bf16* __restrict__ Q, const bf16* __restrict__ Kg,
    const bf16* __restrict__ Vg, bf16* __restrict__ Ctx)
{
    const int S = 2048, H = 2048, DH = 128;
    __shared__ __align__(16) bf16 Ks[64][128];      // 16 KB
    __shared__ __align__(16) bf16 Vt[128][64];      // 16 KB (transposed: [d][k])
    __shared__ __align__(16) bf16 Ps[8][16][64];    // 16 KB (per-wave P patch)

    const int t    = threadIdx.x;
    const int lane = t & 31;
    const int w    = t >> 5;
    const int qt   = blockIdx.x;
    const int h    = blockIdx.y;
    const int b    = blockIdx.z;
    const int qbase = qt * 128;
    const int wrow  = qbase + w * 16;
    const size_t headoff = (size_t)b * S * H + (size_t)h * DH;

    const int rhalf = (lane >> 4) << 3;  // 0 or 8
    const int cl    = lane & 15;
    const float scale = 0.08838834764f;  // 1/sqrt(128)

    // Resident Q fragments covering d = 0..127 (4 x 16x32)
    v16bf qf[4];
    {
        const bf16* qp = Q + headoff + (size_t)wrow * H;
#pragma unroll
        for (int kk = 0; kk < 4; ++kk) qf[kk] = frag_ld_A(qp + kk * 32, H, lane);
    }

    v8f o[8];
    float m_i[8], l_i[8];
#pragma unroll
    for (int d = 0; d < 8; ++d) o[d] = zero8();
#pragma unroll
    for (int r = 0; r < 8; ++r) { m_i[r] = -3e38f; l_i[r] = 0.f; }

    const int nkt = 2 * (qt + 1);  // causal pruning: k tiles of 64 rows
    for (int kt = 0; kt < nkt; ++kt) {
        const int kbase = kt * 64;
        __syncthreads();

#if HAVE_TDM
        // wave 0 kicks the Tensor Data Mover for the 64x128 bf16 K tile
        if (w == 0) {
            uint64_t ga = (uint64_t)(uintptr_t)(Kg + headoff + (size_t)kbase * H);
            uint32_t la = (uint32_t)(uintptr_t)&Ks[0][0];  // LDS byte offset
            uint32x4 g0;
            g0[0] = 1u;                                  // count=1, user mode
            g0[1] = la;                                  // lds_addr
            g0[2] = (uint32_t)ga;                        // global_addr[31:0]
            g0[3] = (uint32_t)((ga >> 32) & 0x01FFFFFFu) | (2u << 30);  // addr[56:32], type=2
            int32x8 g1;
            g1[0] = (int)(1u << 16);       // wg_mask=0, data_size=1 (2 bytes)
            g1[1] = (int)(128u << 16);     // tensor_dim0 = 128 (row length, elements)
            g1[2] = (int)(2048u << 16);    // tensor_dim1 = 2048 (rows)
            g1[3] = (int)(128u << 16);     // tile_dim0 = 128
            g1[4] = (int)64;               // tile_dim1 = 64, tile_dim2 = 0
            g1[5] = (int)2048;             // tensor_dim0_stride = 2048 elements
            g1[6] = 0;                     // stride hi / dim1_stride lo
            g1[7] = 0;
            int32x4 z4 = (int32x4){0, 0, 0, 0};
            int32x8 z8 = (int32x8){0, 0, 0, 0, 0, 0, 0, 0};
            __builtin_amdgcn_tensor_load_to_lds(g0, g1, z4, z4, z8, 0);
        }
#endif
        // all 256 lanes stage V transposed while the TDM runs
#pragma unroll
        for (int i = 0; i < 4; ++i) {
            int c  = t + i * 256;
            int kr = c >> 4;
            int dc = (c & 15) << 3;
#if !HAVE_TDM
            const bf16* kp = Kg + headoff + (size_t)(kbase + kr) * H + dc;
            *(u32x4*)&Ks[kr][dc] = *(const u32x4*)kp;
#endif
            const bf16* vp = Vg + headoff + (size_t)(kbase + kr) * H + dc;
            union { u32x4 q; bf16 e[8]; } uv;
            uv.q = *(const u32x4*)vp;
#pragma unroll
            for (int j = 0; j < 8; ++j) Vt[dc + j][kr] = uv.e[j];
        }
#if HAVE_TDM
        if (w == 0) __builtin_amdgcn_s_wait_tensorcnt(0);
#endif
        __syncthreads();

        // S = Q K^T : 4 n-tiles x 4 K-chunks
        v8f sacc[4];
#pragma unroll
        for (int nt = 0; nt < 4; ++nt) {
            sacc[nt] = zero8();
#pragma unroll
            for (int kk = 0; kk < 4; ++kk) {
                v16bf bk = frag_ld_B(&Ks[nt * 16][kk * 32], 128, lane);
                sacc[nt] = WMMA_BF16(qf[kk], bk, sacc[nt]);
            }
        }

        // scale + causal mask
#pragma unroll
        for (int nt = 0; nt < 4; ++nt) {
            int col = kbase + nt * 16 + cl;
#pragma unroll
            for (int r = 0; r < 8; ++r) {
                int row = wrow + r + rhalf;
                float sv = sacc[nt][r] * scale;
                sacc[nt][r] = (col <= row) ? sv : -3e38f;
            }
        }

        // online softmax: row stats live in 16-lane halves
        float mnew[8], corr[8];
#pragma unroll
        for (int r = 0; r < 8; ++r) {
            float mx = sacc[0][r];
#pragma unroll
            for (int nt = 1; nt < 4; ++nt) mx = fmaxf(mx, sacc[nt][r]);
            for (int off = 1; off < 16; off <<= 1)
                mx = fmaxf(mx, __shfl_xor(mx, off, 32));
            mnew[r] = fmaxf(m_i[r], mx);
            corr[r] = __expf(m_i[r] - mnew[r]);
            m_i[r]  = mnew[r];
        }
#pragma unroll
        for (int r = 0; r < 8; ++r) {
            float rs = 0.f;
#pragma unroll
            for (int nt = 0; nt < 4; ++nt) {
                float p = __expf(sacc[nt][r] - mnew[r]);
                rs += p;
                Ps[w][r + rhalf][nt * 16 + cl] = (bf16)p;
            }
            for (int off = 1; off < 16; off <<= 1)
                rs += __shfl_xor(rs, off, 32);
            l_i[r] = l_i[r] * corr[r] + rs;
        }
#pragma unroll
        for (int dt = 0; dt < 8; ++dt)
#pragma unroll
            for (int r = 0; r < 8; ++r) o[dt][r] *= corr[r];

        // O += P V : 8 d-tiles x 2 K-chunks (Ps wave-private; DScnt waits cover it)
#pragma unroll
        for (int dt = 0; dt < 8; ++dt) {
#pragma unroll
            for (int kk = 0; kk < 2; ++kk) {
                v16bf ap = frag_ld_A(&Ps[w][0][kk * 32], 64, lane);
                v16bf bv = frag_ld_B(&Vt[dt * 16][kk * 32], 64, lane);
                o[dt] = WMMA_BF16(ap, bv, o[dt]);
            }
        }
    }

    // epilogue: normalize and store context as bf16 [N,S,nh,dh]
#pragma unroll
    for (int dt = 0; dt < 8; ++dt) {
#pragma unroll
        for (int r = 0; r < 8; ++r) {
            int row = wrow + r + rhalf;
            int d   = dt * 16 + cl;
            float v = o[dt][r] / l_i[r];
            Ctx[(size_t)b * S * H + (size_t)row * H + (size_t)h * DH + d] = (bf16)v;
        }
    }
}

// ---------------------------------------------------------------------------
// Launch
// ---------------------------------------------------------------------------
extern "C" void kernel_launch(void* const* d_in, const int* in_sizes, int n_in,
                              void* d_out, int out_size, void* d_ws, size_t ws_size,
                              hipStream_t stream) {
    const float* X   = (const float*)d_in[0];
    const int*   pos = (const int*)d_in[1];
    // d_in[2] = mask: causality applied analytically
    const float* Wq  = (const float*)d_in[3];
    const float* Wk  = (const float*)d_in[4];
    const float* Wv  = (const float*)d_in[5];
    const float* Wo  = (const float*)d_in[6];
    const float* bo  = (const float*)d_in[7];
    float* out = (float*)d_out;

    const size_t MB = 1024u * 1024u;
    char* ws = (char*)d_ws;
    bf16* Xh  = (bf16*)(ws + 0);        // 16 MiB : [4096,2048]
    bf16* Wqh = (bf16*)(ws + 16 * MB);  //  8 MiB
    bf16* Wkh = (bf16*)(ws + 24 * MB);
    bf16* Wvh = (bf16*)(ws + 32 * MB);
    bf16* Woh = (bf16*)(ws + 40 * MB);
    bf16* Qh  = (bf16*)(ws + 48 * MB);  // 16 MiB : [2,2048,16,128]
    bf16* Kh  = (bf16*)(ws + 64 * MB);
    bf16* Vh  = (bf16*)(ws + 80 * MB);
    bf16* Ch  = (bf16*)(ws + 96 * MB);  // 16 MiB : context

    const int M = 4096, N = 2048, K = 2048;

    f32_to_bf16_kernel<<<2048, 256, 0, stream>>>(X,  Xh,  (long)M * K);
    f32_to_bf16_kernel<<<1024, 256, 0, stream>>>(Wq, Wqh, (long)N * K);
    f32_to_bf16_kernel<<<1024, 256, 0, stream>>>(Wk, Wkh, (long)N * K);
    f32_to_bf16_kernel<<<1024, 256, 0, stream>>>(Wv, Wvh, (long)N * K);
    f32_to_bf16_kernel<<<1024, 256, 0, stream>>>(Wo, Woh, (long)N * K);

    dim3 gg(N / 128, M / 128);  // (16, 32)
    gemm_bf16_kernel<true, false><<<gg, 256, 0, stream>>>(Xh, Wqh, nullptr, nullptr, Qh, M, N, K);
    gemm_bf16_kernel<true, false><<<gg, 256, 0, stream>>>(Xh, Wkh, nullptr, nullptr, Kh, M, N, K);
    gemm_bf16_kernel<true, false><<<gg, 256, 0, stream>>>(Xh, Wvh, nullptr, nullptr, Vh, M, N, K);

    rope_kernel<<<2048, 256, 0, stream>>>(Qh, Kh, pos, 2L * 2048 * 16 * 64);

    dim3 ga(16, 16, 2);  // (q-tiles, heads, batch)
    flash_attn_kernel<<<ga, 256, 0, stream>>>(Qh, Kh, Vh, Ch);

    gemm_bf16_kernel<false, true><<<gg, 256, 0, stream>>>(Ch, Woh, bo, out, nullptr, M, N, K);
}
